// SPyNetAlignment_47863115546771
// MI455X (gfx1250) — matrix operand order repaired and checked by reading, not compiled, not run on hardware
//
#include <hip/hip_runtime.h>

// ===========================================================================
// EDVR-style pyramid deformable alignment for MI455X (gfx1250, wave32, WMMA).
//
// All GEMM-shaped work (3x3 convs as implicit GEMM, DCN einsum) runs through
// v_wmma_f32_16x16x32_f16: fp32 activations are converted to f16 while being
// staged into LDS in the CDNA5 A-fragment layout; weights are pre-packed once
// into the B-fragment layout; accumulation is fp32 (WMMA C/D).
//
// PT (pixel tiles per block) = 2 for single-source convs & DCN: two
// accumulator chains share each B fragment -> half the L2 weight traffic and
// double WMMA density per loop body.
//
// Input order assumption (setup_inputs dict insertion order, nested dicts in
// insertion order):
//   0:x  1:a3_w 2:a3_b 3:b3_w 4:b3_b 5:a2_w 6:a2_b 7:b2_w 8:b2_b 9:c2_w
//   10:c2_b 11:a1_w 12:a1_b 13:b1_w 14:b1_b 15:c1_w 16:c1_b
//   17:dcn2.ow 18:dcn2.ob 19:dcn2.w 20:dcn2.b
//   21:dcn1.ow 22:dcn1.ob 23:dcn1.w 24:dcn1.b
//   25:dcnf.ow 26:dcnf.ob 27:dcnf.w 28:dcnf.b
// ===========================================================================

typedef __attribute__((ext_vector_type(16))) _Float16 v16h;
typedef __attribute__((ext_vector_type(8)))  float    v8f;

#define BB 2
#define TT 5
#define CC 64
#define H1 128
#define W1 128
#define HW1 (H1*W1)
#define HW2 (64*64)
#define HW3 (32*32)
#define NEG_SLOPE 0.1f

__device__ __forceinline__ v8f wmma16(v16h a, v16h b, v8f c) {
  return __builtin_amdgcn_wmma_f32_16x16x32_f16(false, a, false, b,
                                                (short)0, c, false, false);
}

// Map a K index within a 32-chunk to (half,e) of the ISA 16-bit A/B fragment.
//   r in [0,16):  K = e + half*8            (e in 0..7)
//   r in [16,32): K = (e-8) + 16 + half*8   (e in 8..15)
__device__ __forceinline__ void a_frag_slot(int r, int& hf, int& e) {
  if (r < 16) { hf = r >> 3; e = r & 7; }
  else        { int q = r - 16; hf = q >> 3; e = (q & 7) + 8; }
}

// ---------------------------------------------------------------------------
// Pack fp32 weights (Cout x Ktot, flat K = ci*9 + ky*3 + kx) into the WMMA
// B-fragment tiled layout. Cout zero-padded to a multiple of 16.
// ---------------------------------------------------------------------------
__global__ void pack_weights_kernel(const float* __restrict__ w,
                                    _Float16* __restrict__ wpk,
                                    int Cout, int Ktot) {
  const int nkc  = Ktot / 32;
  const int ngrp = (Cout + 15) >> 4;
  const long total = (long)ngrp * nkc * 512;
  for (long p = (long)blockIdx.x * blockDim.x + threadIdx.x; p < total;
       p += (long)gridDim.x * blockDim.x) {
    int e    = (int)(p & 15);
    int hf   = (int)((p >> 4) & 1);
    int n    = (int)((p >> 5) & 15);
    long q   = p >> 9;            // grp*nkc + kc
    int kc   = (int)(q % nkc);
    int grp  = (int)(q / nkc);
    int oc   = grp * 16 + n;
    int K    = kc * 32 + ((e < 8) ? (hf * 8 + e) : (16 + hf * 8 + (e - 8)));
    float v  = (oc < Cout) ? w[(long)oc * Ktot + K] : 0.f;
    wpk[p] = (_Float16)v;
  }
}

// ---------------------------------------------------------------------------
// 3x3 SAME conv as implicit GEMM with WMMA. PT = 16-pixel tiles per block.
// Block = 128 threads (4 waves); each wave owns one 16-out-channel group and
// PT accumulators sharing every B fragment.
// ---------------------------------------------------------------------------
template <int PT>
__global__ __launch_bounds__(128)
void conv3x3_wmma_kernel(const float* __restrict__ s0,
                         const float* __restrict__ s1,
                         const float* __restrict__ s2,
                         const _Float16* __restrict__ wpk,
                         const float* __restrict__ bias,
                         const float* __restrict__ resid,
                         float* __restrict__ out,
                         int H, int W, int nsrc, int Cout, int act) {
  extern __shared__ _Float16 sA[];   // PT tiles of (16 pixels x Ktot)
  const int Cin  = nsrc * 64;
  const int Ktot = Cin * 9;
  const int nkc  = Ktot >> 5;
  const int tilesPerRow = W / (16 * PT);
  const int h0 = blockIdx.x / tilesPerRow;
  const int w0 = (blockIdx.x % tilesPerRow) * (16 * PT);
  const int tid = threadIdx.x;
  const int HW = H * W;

  // ---- stage im2col tile(s) into LDS (fp32 -> f16, A-fragment layout) ----
  const int total = (Ktot << 4) * PT;
  for (int idx = tid; idx < total; idx += 128) {
    int mp = idx % (16 * PT);     // pixel within block
    int K  = idx / (16 * PT);
    int t  = mp >> 4;             // which 16-pixel tile
    int m  = mp & 15;             // row M within tile
    int ci = K / 9;
    int kk = K - ci * 9;
    int ky = kk / 3 - 1;
    int kx = kk - (kk / 3) * 3 - 1;
    int hh = h0 + ky;
    int ww = w0 + mp + kx;
    float v = 0.f;
    if ((unsigned)hh < (unsigned)H && (unsigned)ww < (unsigned)W) {
      const float* s = (ci < 64) ? s0 : ((ci < 128) ? s1 : s2);
      v = s[(ci & 63) * HW + hh * W + ww];
    }
    int kc = K >> 5, hf, e;
    a_frag_slot(K & 31, hf, e);
    sA[t * (Ktot << 4) + ((kc << 5) + (hf << 4) + m) * 16 + e] = (_Float16)v;
  }
  __syncthreads();

  const int lane = tid & 31;
  const int wave = tid >> 5;
  const int ngrp = (Cout + 15) >> 4;
  const int grp  = blockIdx.y * 4 + wave;
  if (grp >= ngrp) return;   // wave-uniform: EXEC stays all-ones for WMMA

  const int bl = ((lane & 15) << 1) | (lane >> 4);   // B-fragment lane slot
  const _Float16* wg  = wpk + ((long)(grp * nkc) * 32 + bl) * 16;
  const _Float16* ap0 = sA + lane * 16;
  const _Float16* ap1 = sA + (Ktot << 4) + lane * 16;

  v8f c0 = {}, c1 = {};
  for (int kc = 0; kc < nkc; kc += 2) {
    __builtin_prefetch(wg + (long)(kc + 2) * 512, 0, 1);
    v16h b0 = *(const v16h*)(wg + (long)kc * 512);
    v16h b1 = *(const v16h*)(wg + (long)(kc + 1) * 512);
    v16h a00 = *(const v16h*)(ap0 + (long)kc * 512);
    v16h a01 = *(const v16h*)(ap0 + (long)(kc + 1) * 512);
    c0 = wmma16(a00, b0, c0);
    if (PT == 2) {
      v16h a10 = *(const v16h*)(ap1 + (long)kc * 512);
      v16h a11 = *(const v16h*)(ap1 + (long)(kc + 1) * 512);
      c1 = wmma16(a10, b0, c1);
      c0 = wmma16(a01, b1, c0);
      c1 = wmma16(a11, b1, c1);
    } else {
      c0 = wmma16(a01, b1, c0);
    }
  }

  // ---- epilogue: D layout lane&15 = pixel col N, VGPR r -> M = r+(lane>>4)*8
  const int n   = lane & 15;
  const int mh  = (lane >> 4) << 3;
  const int pix = h0 * W + w0 + n;
#pragma unroll
  for (int r8 = 0; r8 < 8; ++r8) {
    int oc = (grp << 4) + mh + r8;
    if (oc < Cout) {
      float bse = bias[oc];
      {
        float v = c0[r8] + bse;
        if (act) v = (v > 0.f) ? v : v * NEG_SLOPE;
        if (resid) v += resid[oc * HW + pix];
        out[oc * HW + pix] = v;
      }
      if (PT == 2) {
        float v = c1[r8] + bse;
        if (act) v = (v > 0.f) ? v : v * NEG_SLOPE;
        if (resid) v += resid[oc * HW + pix + 16];
        out[oc * HW + pix + 16] = v;
      }
    }
  }
}

// ---------------------------------------------------------------------------
// Modulated deformable conv (G=8, 3x3, Cg=8) as WMMA GEMM over K=576.
// 32-pixel tile per block, dual accumulators sharing B fragments.
// co = 216-ch conv out: ch g*18+2k = dy, g*18+2k+1 = dx, 144+g*9+k = mask.
// ---------------------------------------------------------------------------
__global__ __launch_bounds__(128)
void mdcn_wmma_kernel(const float* __restrict__ feat,
                      const float* __restrict__ co,
                      const _Float16* __restrict__ wpk,
                      const float* __restrict__ bias,
                      float* __restrict__ out,
                      int H, int W) {
  __shared__ _Float16 sA[32 * 576];   // two 16-pixel A tiles
  const int HW = H * W;
  const int tilesPerRow = W >> 5;
  const int h0 = blockIdx.x / tilesPerRow;
  const int w0 = (blockIdx.x % tilesPerRow) << 5;
  const int tid = threadIdx.x;

  for (int idx = tid; idx < 32 * 576; idx += 128) {
    int mp  = idx & 31;
    int K   = idx >> 5;
    int t   = mp >> 4;
    int m   = mp & 15;
    int g   = K / 72;
    int rem = K - g * 72;
    int cg  = rem / 9;
    int kk  = rem - cg * 9;
    int pix = h0 * W + w0 + mp;

    float dy  = co[(g * 18 + kk * 2) * HW + pix];
    float dx  = co[(g * 18 + kk * 2 + 1) * HW + pix];
    float ml  = co[(144 + g * 9 + kk) * HW + pix];
    float msk = 1.f / (1.f + __expf(-ml));

    float py = (float)h0 + (float)(kk / 3 - 1) + dy;
    float px = (float)(w0 + mp) + (float)(kk % 3 - 1) + dx;
    float fy = floorf(py), fx = floorf(px);
    int y0 = (int)fy, x0 = (int)fx;
    float wy = py - fy, wx = px - fx;

    const float* f = feat + (long)(g * 8 + cg) * HW;
    float v = 0.f;
#pragma unroll
    for (int jy = 0; jy < 2; ++jy) {
#pragma unroll
      for (int jx = 0; jx < 2; ++jx) {
        int yy = y0 + jy, xx = x0 + jx;
        if ((unsigned)yy < (unsigned)H && (unsigned)xx < (unsigned)W) {
          float wgt = (jy ? wy : 1.f - wy) * (jx ? wx : 1.f - wx);
          v += wgt * f[yy * W + xx];
        }
      }
    }
    v *= msk;

    int kc = K >> 5, hf, e;
    a_frag_slot(K & 31, hf, e);
    sA[t * (576 * 16) + ((kc << 5) + (hf << 4) + m) * 16 + e] = (_Float16)v;
  }
  __syncthreads();

  const int lane = tid & 31;
  const int grp  = tid >> 5;     // 4 waves x 16 out-ch = 64
  const int bl   = ((lane & 15) << 1) | (lane >> 4);
  const _Float16* wg  = wpk + ((long)(grp * 18) * 32 + bl) * 16;
  const _Float16* ap0 = sA + lane * 16;
  const _Float16* ap1 = sA + 576 * 16 + lane * 16;

  v8f c0 = {}, c1 = {};
#pragma unroll
  for (int kc = 0; kc < 18; kc += 2) {
    v16h b0 = *(const v16h*)(wg + kc * 512);
    v16h b1 = *(const v16h*)(wg + (kc + 1) * 512);
    v16h a00 = *(const v16h*)(ap0 + kc * 512);
    v16h a01 = *(const v16h*)(ap0 + (kc + 1) * 512);
    v16h a10 = *(const v16h*)(ap1 + kc * 512);
    v16h a11 = *(const v16h*)(ap1 + (kc + 1) * 512);
    c0 = wmma16(a00, b0, c0);
    c1 = wmma16(a10, b0, c1);
    c0 = wmma16(a01, b1, c0);
    c1 = wmma16(a11, b1, c1);
  }

  const int n   = lane & 15;
  const int mh  = (lane >> 4) << 3;
  const int pix = h0 * W + w0 + n;
#pragma unroll
  for (int r8 = 0; r8 < 8; ++r8) {
    int oc = (grp << 4) + mh + r8;
    float bse = bias[oc];
    out[oc * HW + pix]      = c0[r8] + bse;
    out[oc * HW + pix + 16] = c1[r8] + bse;
  }
}

// ---------------------------------------------------------------------------
// Elementwise helpers
// ---------------------------------------------------------------------------
__global__ void down2_kernel(const float* __restrict__ in, float* __restrict__ out,
                             int N, int Hi, int Wi) {
  const int Ho = Hi >> 1, Wo = Wi >> 1;
  const long total = (long)N * Ho * Wo;
  for (long idx = (long)blockIdx.x * blockDim.x + threadIdx.x; idx < total;
       idx += (long)gridDim.x * blockDim.x) {
    int n = (int)(idx / (Ho * Wo));
    int r = (int)(idx % (Ho * Wo));
    int i = r / Wo, j = r % Wo;
    const float* p = in + (long)n * Hi * Wi + (2 * i) * Wi + 2 * j;
    out[idx] = 0.25f * (p[0] + p[1] + p[Wi] + p[Wi + 1]);
  }
}

__global__ void upsample2x_kernel(const float* __restrict__ in, float* __restrict__ out,
                                  int Hi, int Wi) {
  const int Ho = Hi * 2, Wo = Wi * 2;
  const long total = (long)CC * Ho * Wo;
  for (long idx = (long)blockIdx.x * blockDim.x + threadIdx.x; idx < total;
       idx += (long)gridDim.x * blockDim.x) {
    int ch = (int)(idx / (Ho * Wo));
    int r  = (int)(idx % (Ho * Wo));
    int i = r / Wo, j = r % Wo;
    float sy = (i + 0.5f) * 0.5f - 0.5f;
    float sx = (j + 0.5f) * 0.5f - 0.5f;
    float fy = floorf(sy), fx = floorf(sx);
    int y0 = (int)fy, x0 = (int)fx;
    float wy = sy - fy, wx = sx - fx;
    int y0c = min(max(y0, 0), Hi - 1), y1c = min(max(y0 + 1, 0), Hi - 1);
    int x0c = min(max(x0, 0), Wi - 1), x1c = min(max(x0 + 1, 0), Wi - 1);
    const float* p = in + (long)ch * Hi * Wi;
    float v = (1.f - wy) * ((1.f - wx) * p[y0c * Wi + x0c] + wx * p[y0c * Wi + x1c]) +
              wy        * ((1.f - wx) * p[y1c * Wi + x0c] + wx * p[y1c * Wi + x1c]);
    out[idx] = v * 2.f;    // reference does up2(x) * 2
  }
}

__global__ void copy_kernel(const float* __restrict__ in, float* __restrict__ out, long n) {
  for (long i = (long)blockIdx.x * blockDim.x + threadIdx.x; i < n;
       i += (long)gridDim.x * blockDim.x)
    out[i] = in[i];
}

// ---------------------------------------------------------------------------
// Host-side launch helpers
// ---------------------------------------------------------------------------
static void launch_conv(const float* s0, const float* s1, const float* s2,
                        const _Float16* wpk, const float* bias, const float* resid,
                        float* out, int H, int W, int nsrc, int Cout, int act,
                        hipStream_t stream) {
  int ngrp = (Cout + 15) / 16;
  if (nsrc == 1) {   // PT=2: 32-pixel tiles, 36 KB LDS
    dim3 grid((H * W) / 32, (ngrp + 3) / 4);
    size_t smem = (size_t)32 * 64 * 9 * sizeof(_Float16);
    conv3x3_wmma_kernel<2><<<grid, 128, smem, stream>>>(s0, s1, s2, wpk, bias,
                                                        resid, out, H, W, nsrc,
                                                        Cout, act);
  } else {           // PT=1 for concat inputs (36/55 KB LDS already)
    dim3 grid((H * W) / 16, (ngrp + 3) / 4);
    size_t smem = (size_t)16 * nsrc * 64 * 9 * sizeof(_Float16);
    conv3x3_wmma_kernel<1><<<grid, 128, smem, stream>>>(s0, s1, s2, wpk, bias,
                                                        resid, out, H, W, nsrc,
                                                        Cout, act);
  }
}

static void launch_pack(const float* w, _Float16* wpk, int Cout, int Ktot,
                        hipStream_t stream) {
  pack_weights_kernel<<<128, 256, 0, stream>>>(w, wpk, Cout, Ktot);
}

extern "C" void kernel_launch(void* const* d_in, const int* in_sizes, int n_in,
                              void* d_out, int out_size, void* d_ws, size_t ws_size,
                              hipStream_t stream) {
  const float* x = (const float*)d_in[0];
  const float* a3_w = (const float*)d_in[1];  const float* a3_b = (const float*)d_in[2];
  const float* b3_w = (const float*)d_in[3];  const float* b3_b = (const float*)d_in[4];
  const float* a2_w = (const float*)d_in[5];  const float* a2_b = (const float*)d_in[6];
  const float* b2_w = (const float*)d_in[7];  const float* b2_b = (const float*)d_in[8];
  const float* c2_w = (const float*)d_in[9];  const float* c2_b = (const float*)d_in[10];
  const float* a1_w = (const float*)d_in[11]; const float* a1_b = (const float*)d_in[12];
  const float* b1_w = (const float*)d_in[13]; const float* b1_b = (const float*)d_in[14];
  const float* c1_w = (const float*)d_in[15]; const float* c1_b = (const float*)d_in[16];
  const float* d2_ow = (const float*)d_in[17]; const float* d2_ob = (const float*)d_in[18];
  const float* d2_w  = (const float*)d_in[19]; const float* d2_b  = (const float*)d_in[20];
  const float* d1_ow = (const float*)d_in[21]; const float* d1_ob = (const float*)d_in[22];
  const float* d1_w  = (const float*)d_in[23]; const float* d1_b  = (const float*)d_in[24];
  const float* df_ow = (const float*)d_in[25]; const float* df_ob = (const float*)d_in[26];
  const float* df_w  = (const float*)d_in[27]; const float* df_b  = (const float*)d_in[28];
  float* out = (float*)d_out;

  // ---- workspace layout (floats) ----
  float* ws = (float*)d_ws;
  size_t o = 0;
  float* x2 = ws + o;     o += (size_t)BB * TT * CC * HW2;
  float* x3 = ws + o;     o += (size_t)BB * TT * CC * HW3;
  float* t_o3a  = ws + o; o += (size_t)CC * HW3;
  float* t_o3   = ws + o; o += (size_t)CC * HW3;
  float* t_uo3  = ws + o; o += (size_t)CC * HW2;
  float* t_off2 = ws + o; o += (size_t)216 * HW2;
  float* t_fa2  = ws + o; o += (size_t)CC * HW2;
  float* t_o2a  = ws + o; o += (size_t)CC * HW2;
  float* t_o2b  = ws + o; o += (size_t)CC * HW2;
  float* t_o2   = ws + o; o += (size_t)CC * HW2;
  float* t_uo2  = ws + o; o += (size_t)CC * HW1;
  float* t_off1 = ws + o; o += (size_t)216 * HW1;
  float* t_fa1  = ws + o; o += (size_t)CC * HW1;
  float* t_o1a  = ws + o; o += (size_t)CC * HW1;
  float* t_o1b  = ws + o; o += (size_t)CC * HW1;
  float* t_o1   = ws + o; o += (size_t)CC * HW1;

  _Float16* wp = (_Float16*)(ws + o);
  size_t wo = 0;
  _Float16* wp_a3 = wp + wo; wo += (size_t)64 * 1152;
  _Float16* wp_b3 = wp + wo; wo += (size_t)64 * 576;
  _Float16* wp_a2 = wp + wo; wo += (size_t)64 * 1728;
  _Float16* wp_b2 = wp + wo; wo += (size_t)64 * 576;
  _Float16* wp_c2 = wp + wo; wo += (size_t)64 * 576;
  _Float16* wp_a1 = wp + wo; wo += (size_t)64 * 1728;
  _Float16* wp_b1 = wp + wo; wo += (size_t)64 * 576;
  _Float16* wp_c1 = wp + wo; wo += (size_t)64 * 576;
  _Float16* wp_ow2 = wp + wo; wo += (size_t)224 * 576;   // Cout 216 padded to 224
  _Float16* wp_ow1 = wp + wo; wo += (size_t)224 * 576;
  _Float16* wp_owf = wp + wo; wo += (size_t)224 * 576;
  _Float16* wp_w2 = wp + wo; wo += (size_t)64 * 576;
  _Float16* wp_w1 = wp + wo; wo += (size_t)64 * 576;
  _Float16* wp_wf = wp + wo; wo += (size_t)64 * 576;

  // ---- 1. pack all weights to f16 B-fragment layout ----
  launch_pack(a3_w, wp_a3, 64, 1152, stream);
  launch_pack(b3_w, wp_b3, 64, 576,  stream);
  launch_pack(a2_w, wp_a2, 64, 1728, stream);
  launch_pack(b2_w, wp_b2, 64, 576,  stream);
  launch_pack(c2_w, wp_c2, 64, 576,  stream);
  launch_pack(a1_w, wp_a1, 64, 1728, stream);
  launch_pack(b1_w, wp_b1, 64, 576,  stream);
  launch_pack(c1_w, wp_c1, 64, 576,  stream);
  launch_pack(d2_ow, wp_ow2, 216, 576, stream);
  launch_pack(d1_ow, wp_ow1, 216, 576, stream);
  launch_pack(df_ow, wp_owf, 216, 576, stream);
  launch_pack(d2_w, wp_w2, 64, 576, stream);
  launch_pack(d1_w, wp_w1, 64, 576, stream);
  launch_pack(df_w, wp_wf, 64, 576, stream);

  // ---- 2. pyramid downsample ----
  down2_kernel<<<1024, 256, 0, stream>>>(x, x2, BB * TT * CC, H1, W1);
  down2_kernel<<<512, 256, 0, stream>>>(x2, x3, BB * TT * CC, 64, 64);

  // ---- 3. passthrough of center frame ----
  for (int b = 0; b < BB; ++b) {
    long off = ((long)b * TT + 2) * CC * HW1;
    copy_kernel<<<512, 256, 0, stream>>>(x + off, out + off, (long)CC * HW1);
  }

  // ---- 4. per (batch, non-center frame) pipeline ----
  for (int b = 0; b < BB; ++b) {
    const float* fc1 = x  + ((long)b * TT + 2) * CC * HW1;
    const float* fc2 = x2 + ((long)b * TT + 2) * CC * HW2;
    const float* fc3 = x3 + ((long)b * TT + 2) * CC * HW3;
    for (int fi = 0; fi < 4; ++fi) {
      int ii = (fi < 2) ? fi : fi + 1;     // 0,1,3,4 (skip center 2)
      const float* x1_bi = x  + ((long)b * TT + ii) * CC * HW1;
      const float* x2_bi = x2 + ((long)b * TT + ii) * CC * HW2;
      const float* x3_bi = x3 + ((long)b * TT + ii) * CC * HW3;
      float* out_bi = out + ((long)b * TT + ii) * CC * HW1;

      // level 3
      launch_conv(x3_bi, fc3, nullptr, wp_a3, a3_b, nullptr, t_o3a, 32, 32, 2, 64, 1, stream);
      launch_conv(t_o3a, nullptr, nullptr, wp_b3, b3_b, nullptr, t_o3, 32, 32, 1, 64, 0, stream);
      upsample2x_kernel<<<512, 256, 0, stream>>>(t_o3, t_uo3, 32, 32);

      // level 2
      launch_conv(t_uo3, nullptr, nullptr, wp_ow2, d2_ob, nullptr, t_off2, 64, 64, 1, 216, 0, stream);
      mdcn_wmma_kernel<<<HW2 / 32, 128, 0, stream>>>(x2_bi, t_off2, wp_w2, d2_b, t_fa2, 64, 64);
      launch_conv(t_fa2, fc2, t_uo3, wp_a2, a2_b, nullptr, t_o2a, 64, 64, 3, 64, 1, stream);
      launch_conv(t_o2a, nullptr, nullptr, wp_b2, b2_b, nullptr, t_o2b, 64, 64, 1, 64, 1, stream);
      launch_conv(t_o2b, nullptr, nullptr, wp_c2, c2_b, t_uo3, t_o2, 64, 64, 1, 64, 0, stream);
      upsample2x_kernel<<<1024, 256, 0, stream>>>(t_o2, t_uo2, 64, 64);

      // level 1
      launch_conv(t_uo2, nullptr, nullptr, wp_ow1, d1_ob, nullptr, t_off1, 128, 128, 1, 216, 0, stream);
      mdcn_wmma_kernel<<<HW1 / 32, 128, 0, stream>>>(x1_bi, t_off1, wp_w1, d1_b, t_fa1, 128, 128);
      launch_conv(t_fa1, fc1, t_uo2, wp_a1, a1_b, nullptr, t_o1a, 128, 128, 3, 64, 1, stream);
      launch_conv(t_o1a, nullptr, nullptr, wp_b1, b1_b, nullptr, t_o1b, 128, 128, 1, 64, 1, stream);
      launch_conv(t_o1b, nullptr, nullptr, wp_c1, c1_b, t_uo2, t_o1, 128, 128, 1, 64, 0, stream);

      // final cascading DCN -> output
      launch_conv(t_o1, nullptr, nullptr, wp_owf, df_ob, nullptr, t_off1, 128, 128, 1, 216, 0, stream);
      mdcn_wmma_kernel<<<HW1 / 32, 128, 0, stream>>>(x1_bi, t_off1, wp_wf, df_b, out_bi, 128, 128);
    }
  }
}
